// FDA_Module_21500606283969
// MI455X (gfx1250) — matrix operand order repaired and checked by reading, not compiled, not run on hardware
//
#include <hip/hip_runtime.h>
#include <hip/hip_bf16.h>

typedef __bf16 bf16_t;
typedef __attribute__((ext_vector_type(16))) __bf16 v16bf;
typedef __attribute__((ext_vector_type(4)))  __bf16 v4bf;
typedef __attribute__((ext_vector_type(8)))  float  v8f;
typedef __attribute__((ext_vector_type(4)))  unsigned int u32x4;
typedef int v4i_t __attribute__((vector_size(16)));   // matches builtin's V4i

union FragBF { u32x4 q[2]; v16bf v; };

#if defined(__has_builtin)
#if __has_builtin(__builtin_amdgcn_global_load_async_to_lds_b128) && \
    __has_builtin(__builtin_amdgcn_s_wait_asynccnt)
#define USE_ASYNC_LDS 1
#endif
#endif
#ifndef USE_ASYNC_LDS
#define USE_ASYNC_LDS 0
#endif

#define AS1_V4I(p) ((__attribute__((address_space(1))) v4i_t*)(p))
#define AS3_V4I(p) ((__attribute__((address_space(3))) v4i_t*)(p))

// ---------------------------------------------------------------------------
// Weight convert + transpose: W (K x N, f32) -> Wt (N x K, bf16)
// ---------------------------------------------------------------------------
__global__ __launch_bounds__(256) void wt_kernel(const float* __restrict__ W,
                                                 bf16_t* __restrict__ Wt,
                                                 int K, int N) {
    int idx = blockIdx.x * 256 + threadIdx.x;
    if (idx >= K * N) return;
    int k = idx / N;
    int n = idx - k * N;
    Wt[(size_t)n * K + k] = (bf16_t)W[idx];
}

// ---------------------------------------------------------------------------
// LayerNorm over D=1024, one wave32 per row, output bf16
// ---------------------------------------------------------------------------
__global__ __launch_bounds__(256) void ln_kernel(const float* __restrict__ x,
                                                 const float* __restrict__ gamma,
                                                 const float* __restrict__ beta,
                                                 bf16_t* __restrict__ out, int M) {
    const int wave = threadIdx.x >> 5;
    const int lane = threadIdx.x & 31;
    const int row  = blockIdx.x * 8 + wave;
    if (row >= M) return;

    const float* xr = x + (size_t)row * 1024;
    float vals[32];
    float s = 0.f, s2 = 0.f;
#pragma unroll
    for (int i = 0; i < 8; ++i) {
        const float4 v = ((const float4*)xr)[lane + i * 32];
        vals[i*4+0] = v.x; vals[i*4+1] = v.y; vals[i*4+2] = v.z; vals[i*4+3] = v.w;
        s  += v.x + v.y + v.z + v.w;
        s2 += v.x*v.x + v.y*v.y + v.z*v.z + v.w*v.w;
    }
#pragma unroll
    for (int off = 16; off > 0; off >>= 1) {
        s  += __shfl_xor(s,  off, 32);
        s2 += __shfl_xor(s2, off, 32);
    }
    const float mu   = s * (1.0f / 1024.0f);
    const float var  = s2 * (1.0f / 1024.0f) - mu * mu;
    const float rstd = rsqrtf(var + 1e-5f);

    bf16_t* outr = out + (size_t)row * 1024;
#pragma unroll
    for (int i = 0; i < 8; ++i) {
        const float4 g = ((const float4*)gamma)[lane + i * 32];
        const float4 b = ((const float4*)beta )[lane + i * 32];
        v4bf o;
        o[0] = (bf16_t)((vals[i*4+0] - mu) * rstd * g.x + b.x);
        o[1] = (bf16_t)((vals[i*4+1] - mu) * rstd * g.y + b.y);
        o[2] = (bf16_t)((vals[i*4+2] - mu) * rstd * g.z + b.z);
        o[3] = (bf16_t)((vals[i*4+3] - mu) * rstd * g.w + b.w);
        ((v4bf*)outr)[lane + i * 32] = o;
    }
}

// ---------------------------------------------------------------------------
// bf16 WMMA GEMM: out = act(A[M,K] x Bt[N,K]^T + bias)  (+ residual for FINAL)
// Block: 256 threads = 8 waves; tile 128(M) x 64(N); wave tile 32x32 (2x2 WMMA)
// B panel staged through LDS (padded stride 40 bf16 -> conflict-free b128).
// Async path: global_load_async_to_lds_b128 + s_wait_asynccnt, LDS dbl-buffer.
// Fallback: register-double-buffered global prefetch + ds_store_b128.
// ---------------------------------------------------------------------------
template <int K, bool GELU, bool FINAL>
__global__ __launch_bounds__(256) void gemm_kernel(
    const bf16_t* __restrict__ A,     // M x K
    const bf16_t* __restrict__ Bt,    // N x K (transposed weights)
    const float*  __restrict__ bias,  // N
    bf16_t*       __restrict__ Obf,   // M x N  (when !FINAL)
    float*        __restrict__ Of,    // M x N  (when FINAL)
    const float*  __restrict__ resid, // M x N  (when FINAL)
    int N) {
    constexpr int LDSS = 40;                  // 32 k-elems + 8 pad (bf16)
    constexpr int BUFE = 64 * LDSS;           // elems per buffer (5 KB)
    __shared__ bf16_t bsh[2 * BUFE];          // double buffer (10 KB)

    const int tid  = threadIdx.x;
    const int wave = tid >> 5;
    const int lane = tid & 31;
    const int wm   = wave & 3;                // 0..3 -> M offset
    const int wn   = wave >> 2;               // 0..1 -> N offset
    const int half = lane >> 4;               // lane half (K-chunk select)
    const int l16  = lane & 15;

    const size_t m0 = (size_t)blockIdx.y * 128;
    const int    n0 = blockIdx.x * 64;

    // B staging: 64 rows x 32 k = 4 KB per k-step, one 16B chunk per thread
    const int srow = tid >> 2;                // 0..63
    const int scol = (tid & 3) * 8;           // 0,8,16,24 (bf16 elems)
    const bf16_t* bsrc = Bt + (size_t)(n0 + srow) * K + scol;
    bf16_t*       bdst = bsh + srow * LDSS + scol;

    // A fragment base pointers (two 16-row subtiles per wave)
    const bf16_t* arow0 = A + (m0 + (size_t)(wm * 32 + l16)) * K + half * 8;
    const bf16_t* arow1 = arow0 + (size_t)16 * K;

    v8f acc[2][2] = {};
    const int nk = K / 32;

#if USE_ASYNC_LDS
    // ---- CDNA5 async staging: ASYNCcnt-tracked direct global->LDS DMA ----
    __builtin_amdgcn_global_load_async_to_lds_b128(AS1_V4I(bsrc), AS3_V4I(bdst), 0, 0);
    for (int ks = 0; ks < nk; ++ks) {
        __builtin_amdgcn_s_wait_asynccnt(0);
        __syncthreads();
        if (ks + 1 < nk)
            __builtin_amdgcn_global_load_async_to_lds_b128(
                AS1_V4I(bsrc + (size_t)(ks + 1) * 32),
                AS3_V4I(bdst + ((ks + 1) & 1) * BUFE), 0, 0);
        const bf16_t* buf = bsh + (ks & 1) * BUFE;
        const int kO = ks * 32;

        FragBF a[2];
        a[0].q[0] = *(const u32x4*)(arow0 + kO);
        a[0].q[1] = *(const u32x4*)(arow0 + kO + 16);
        a[1].q[0] = *(const u32x4*)(arow1 + kO);
        a[1].q[1] = *(const u32x4*)(arow1 + kO + 16);

        FragBF b[2];
#pragma unroll
        for (int j = 0; j < 2; ++j) {
            const bf16_t* bp = buf + (wn * 32 + j * 16 + l16) * LDSS + half * 8;
            b[j].q[0] = *(const u32x4*)bp;
            b[j].q[1] = *(const u32x4*)(bp + 16);
        }

#pragma unroll
        for (int i = 0; i < 2; ++i)
#pragma unroll
            for (int j = 0; j < 2; ++j)
                acc[i][j] = __builtin_amdgcn_wmma_f32_16x16x32_bf16(
                    false, a[i].v, false, b[j].v, (short)0, acc[i][j],
                    false, false);
    }
#else
    // ---- Fallback: register-double-buffered global prefetch + ds_store ----
    u32x4 pre = *(const u32x4*)bsrc;          // prefetch k-step 0
    for (int ks = 0; ks < nk; ++ks) {
        *(u32x4*)bdst = pre;                  // commit staged B chunk
        __syncthreads();
        if (ks + 1 < nk)                      // overlap next fetch with compute
            pre = *(const u32x4*)(bsrc + (size_t)(ks + 1) * 32);

        const int kO = ks * 32;

        FragBF a[2];
        a[0].q[0] = *(const u32x4*)(arow0 + kO);
        a[0].q[1] = *(const u32x4*)(arow0 + kO + 16);
        a[1].q[0] = *(const u32x4*)(arow1 + kO);
        a[1].q[1] = *(const u32x4*)(arow1 + kO + 16);

        FragBF b[2];
#pragma unroll
        for (int j = 0; j < 2; ++j) {
            const bf16_t* bp = bsh + (wn * 32 + j * 16 + l16) * LDSS + half * 8;
            b[j].q[0] = *(const u32x4*)bp;
            b[j].q[1] = *(const u32x4*)(bp + 16);
        }

#pragma unroll
        for (int i = 0; i < 2; ++i)
#pragma unroll
            for (int j = 0; j < 2; ++j)
                acc[i][j] = __builtin_amdgcn_wmma_f32_16x16x32_bf16(
                    false, a[i].v, false, b[j].v, (short)0, acc[i][j],
                    false, false);

        __syncthreads();
    }
#endif

    // Epilogue: bias (+ exact erf GELU) (+ residual). C/D layout:
    // VGPR r -> row (r + 8*half), lane&15 -> col.
#pragma unroll
    for (int j = 0; j < 2; ++j) {
        const int col = n0 + wn * 32 + j * 16 + l16;
        const float bv = bias[col];
#pragma unroll
        for (int i = 0; i < 2; ++i) {
            const size_t rowb = m0 + (size_t)(wm * 32 + i * 16 + half * 8);
#pragma unroll
            for (int r = 0; r < 8; ++r) {
                float v = acc[i][j][r] + bv;
                if (GELU) v = 0.5f * v * (1.0f + erff(v * 0.70710678118654752f));
                if (FINAL) {
                    const size_t off = (rowb + r) * (size_t)N + col;
                    Of[off] = resid[off] + v;
                } else {
                    Obf[(rowb + r) * (size_t)N + col] = (bf16_t)v;
                }
            }
        }
    }
}

// ---------------------------------------------------------------------------
extern "C" void kernel_launch(void* const* d_in, const int* in_sizes, int n_in,
                              void* d_out, int out_size, void* d_ws, size_t ws_size,
                              hipStream_t stream) {
    const float* x     = (const float*)d_in[0];
    const float* gamma = (const float*)d_in[1];
    const float* beta  = (const float*)d_in[2];
    const float* W1 = (const float*)d_in[3];
    const float* b1 = (const float*)d_in[4];
    const float* W2 = (const float*)d_in[5];
    const float* b2 = (const float*)d_in[6];
    const float* W3 = (const float*)d_in[7];
    const float* b3 = (const float*)d_in[8];
    const float* W4 = (const float*)d_in[9];
    const float* b4 = (const float*)d_in[10];

    const int D = 1024, H1 = 512, H2 = 512, H3 = 256;
    const int M = in_sizes[0] / D;            // 32768

    char* ws = (char*)d_ws;
    bf16_t* xln = (bf16_t*)ws; ws += (size_t)M * D  * sizeof(bf16_t);
    bf16_t* h1  = (bf16_t*)ws; ws += (size_t)M * H1 * sizeof(bf16_t);
    bf16_t* h2  = (bf16_t*)ws; ws += (size_t)M * H2 * sizeof(bf16_t);
    bf16_t* h3  = (bf16_t*)ws; ws += (size_t)M * H3 * sizeof(bf16_t);
    bf16_t* W1t = (bf16_t*)ws; ws += (size_t)H1 * D  * sizeof(bf16_t);
    bf16_t* W2t = (bf16_t*)ws; ws += (size_t)H2 * H1 * sizeof(bf16_t);
    bf16_t* W3t = (bf16_t*)ws; ws += (size_t)H3 * H2 * sizeof(bf16_t);
    bf16_t* W4t = (bf16_t*)ws; ws += (size_t)D  * H3 * sizeof(bf16_t);

    // Weight convert+transpose (tiny)
    wt_kernel<<<(D  * H1 + 255) / 256, 256, 0, stream>>>(W1, W1t, D,  H1);
    wt_kernel<<<(H1 * H2 + 255) / 256, 256, 0, stream>>>(W2, W2t, H1, H2);
    wt_kernel<<<(H2 * H3 + 255) / 256, 256, 0, stream>>>(W3, W3t, H2, H3);
    wt_kernel<<<(H3 * D  + 255) / 256, 256, 0, stream>>>(W4, W4t, H3, D);

    // LayerNorm -> bf16
    ln_kernel<<<(M + 7) / 8, 256, 0, stream>>>(x, gamma, beta, xln, M);

    // GEMM chain (bias+GELU fused; residual fused into last)
    gemm_kernel<1024, true,  false><<<dim3(H1 / 64, M / 128), 256, 0, stream>>>(
        xln, W1t, b1, h1, nullptr, nullptr, H1);
    gemm_kernel<512,  true,  false><<<dim3(H2 / 64, M / 128), 256, 0, stream>>>(
        h1, W2t, b2, h2, nullptr, nullptr, H2);
    gemm_kernel<512,  true,  false><<<dim3(H3 / 64, M / 128), 256, 0, stream>>>(
        h2, W3t, b3, h3, nullptr, nullptr, H3);
    gemm_kernel<256,  false, true ><<<dim3(D / 64, M / 128), 256, 0, stream>>>(
        h3, W4t, b4, nullptr, (float*)d_out, x, D);
}
